// MHA_rotary_17403207483619
// MI455X (gfx1250) — matrix operand order repaired and checked by reading, not compiled, not run on hardware
//
#include <hip/hip_runtime.h>
#include <stdint.h>

typedef __bf16 bf16;
typedef __attribute__((ext_vector_type(16))) __bf16 v16bf;
typedef __attribute__((ext_vector_type(8)))  float  v8f;

#define BB 4
#define TT 2048
#define CC 1024
#define HH 16
#define DD 64
#define RR 32

// LDS row pitches (bf16 elements), padded so 16-lane fragment reads hit
// distinct banks (row stride mod 64 dwords = 4 (W) or 36 (K) -> 16 rows x 4 dwords all distinct).
#define WPITCH 136   // GEMM weight chunk rows: 128 data + 8 pad (272B)
#define KPITCH 72    // attention tile rows:    64 data + 8 pad (144B)

union BF16Frag { uint4 u[2]; v16bf v; };

__device__ inline v8f vzero8() {
  v8f z = {0.f, 0.f, 0.f, 0.f, 0.f, 0.f, 0.f, 0.f};
  return z;
}

__device__ inline v8f wmma_bf16(v16bf a, v16bf b, v8f c) {
  return __builtin_amdgcn_wmma_f32_16x16x32_bf16(
      false, a, false, b, (short)0, c, false, false);
}

// Async global->LDS copy (ASYNCcnt path). Flat shared-pointer low 32 bits are
// the LDS byte offset (aperture keeps the offset in addr[31:0]).
__device__ inline void async_load_b128(void* lds_ptr, const void* gptr) {
  uint32_t l = (uint32_t)(uintptr_t)lds_ptr;
  uint64_t g = (uint64_t)(uintptr_t)gptr;
  asm volatile("global_load_async_to_lds_b128 %0, %1, off"
               :: "v"(l), "v"(g) : "memory");
}
__device__ inline void wait_async0() {
  asm volatile("s_wait_asynccnt 0x0" ::: "memory");
}

// A fragment: 16x32 (M x K), CDNA5 16-bit A layout. Row M = lane&15;
// lanes 0-15 hold K = {0..7, 16..23}, lanes 16-31 hold K = {8..15, 24..31}.
__device__ inline v16bf load_a_frag(const bf16* __restrict__ base, int ld, int kbase) {
  const int lane = threadIdx.x & 31;
  const int row  = lane & 15;
  const int k0   = kbase + ((lane >> 4) << 3);
  const bf16* p  = base + (size_t)row * ld + k0;
  BF16Frag f;
  f.u[0] = *(const uint4*)(p);
  f.u[1] = *(const uint4*)(p + 16);
  return f.v;
}

// B fragment: 32x16 (K x N) from an "n-major" source (row n holds consecutive
// k). Lane n = lane&15; lanes 0-15 hold K={0..7,16..23}, lanes 16-31 hold K={8..15,24..31}.
__device__ inline v16bf load_b_frag(const bf16* __restrict__ base, int ld, int nbase, int kbase) {
  const int lane = threadIdx.x & 31;
  const int n    = nbase + (lane & 15);
  const int k0   = kbase + ((lane >> 4) << 3);
  const bf16* p  = base + (size_t)n * ld + k0;
  BF16Frag f;
  f.u[0] = *(const uint4*)(p);
  f.u[1] = *(const uint4*)(p + 16);
  return f.v;
}

// ---------------------------------------------------------------------------
// f32 -> bf16 conversion (vectorized x4)
// ---------------------------------------------------------------------------
__global__ __launch_bounds__(256) void cvt_bf16_kernel(
    const float* __restrict__ in, bf16* __restrict__ out, int n4) {
  int i = blockIdx.x * 256 + threadIdx.x;
  if (i >= n4) return;
  float4 f = ((const float4*)in)[i];
  bf16 o[4] = {(bf16)f.x, (bf16)f.y, (bf16)f.z, (bf16)f.w};
  *(uint2*)(out + (size_t)i * 4) = *(uint2*)o;
}

// ---------------------------------------------------------------------------
// Shared GEMM mainloop, 2 M-tiles per wave (32 rows): every B fragment from
// LDS feeds two WMMAs. Weight chunks (64 N-rows x 128 K) double-buffered in
// LDS via async loads; 4 WMMA K-steps per chunk, one barrier per chunk, next
// chunk prefetched during compute. All fragments of a K-step are loaded
// before the 8-WMMA burst so a single wait covers the whole step.
// ---------------------------------------------------------------------------
__device__ inline void gemm_mainloop2(const bf16* __restrict__ a_base,
                                      const bf16* __restrict__ wb, int n0,
                                      bf16* __restrict__ wl,  // [2][64*WPITCH]
                                      v8f acc[2][4]) {
  const int tid = threadIdx.x;
#pragma unroll
  for (int i = 0; i < 4; ++i) {  // issue chunk 0: 256 thr x 4 x 16B
    int u = tid + 256 * i;
    int row = u >> 4, c = u & 15;
    async_load_b128(&wl[row * WPITCH + c * 8],
                    wb + (size_t)(n0 + row) * CC + c * 8);
  }
  int buf = 0;
  for (int kc = 0; kc < CC; kc += 128) {
    wait_async0();
    __syncthreads();  // chunk[buf] visible to all waves
    bf16* cur = wl + buf * (64 * WPITCH);
    if (kc + 128 < CC) {  // prefetch next chunk into the retired buffer
      bf16* nxt = wl + (buf ^ 1) * (64 * WPITCH);
#pragma unroll
      for (int i = 0; i < 4; ++i) {
        int u = tid + 256 * i;
        int row = u >> 4, c = u & 15;
        async_load_b128(&nxt[row * WPITCH + c * 8],
                        wb + (size_t)(n0 + row) * CC + kc + 128 + c * 8);
      }
    }
#pragma unroll
    for (int ks = 0; ks < 4; ++ks) {
      v16bf a0 = load_a_frag(a_base, CC, kc + ks * 32);
      v16bf a1 = load_a_frag(a_base + 16 * CC, CC, kc + ks * 32);
      v16bf b0 = load_b_frag(cur, WPITCH, 0,  ks * 32);
      v16bf b1 = load_b_frag(cur, WPITCH, 16, ks * 32);
      v16bf b2 = load_b_frag(cur, WPITCH, 32, ks * 32);
      v16bf b3 = load_b_frag(cur, WPITCH, 48, ks * 32);
      acc[0][0] = wmma_bf16(a0, b0, acc[0][0]);
      acc[1][0] = wmma_bf16(a1, b0, acc[1][0]);
      acc[0][1] = wmma_bf16(a0, b1, acc[0][1]);
      acc[1][1] = wmma_bf16(a1, b1, acc[1][1]);
      acc[0][2] = wmma_bf16(a0, b2, acc[0][2]);
      acc[1][2] = wmma_bf16(a1, b2, acc[1][2]);
      acc[0][3] = wmma_bf16(a0, b3, acc[0][3]);
      acc[1][3] = wmma_bf16(a1, b3, acc[1][3]);
    }
    buf ^= 1;
  }
}

// ---------------------------------------------------------------------------
// QKV projection: y = x @ W^T + b (+RoPE, +scale). Block 256 thr (8 waves),
// tile 256(M) x 64(N); the N-tile is exactly one head.
//  store_mode 0: out(B,H,T,D)  (q, k)
//  store_mode 1: out(B,H,D,T)  (v, transposed so attention can async-load
//                V^T tiles straight to LDS; 8 consecutive tokens per lane
//                pack into one 16B store)
// softmax 1/sqrt(D) * log2(e) is folded into the q projection via out_scale.
// ---------------------------------------------------------------------------
__global__ __launch_bounds__(256) void qkv_gemm_kernel(
    const bf16* __restrict__ xb, const bf16* __restrict__ wb,
    const float* __restrict__ bias, const float* __restrict__ rope,
    bf16* __restrict__ out, int do_rope, int store_mode, float out_scale) {
  __shared__ bf16 wl[2][64 * WPITCH];
  const int tid  = threadIdx.x;
  const int wave = tid >> 5, lane = tid & 31;
  const int m0 = blockIdx.x * 256 + wave * 32;
  const int n0 = blockIdx.y * 64;  // == head * 64
  v8f acc[2][4] = {{vzero8(), vzero8(), vzero8(), vzero8()},
                   {vzero8(), vzero8(), vzero8(), vzero8()}};
  gemm_mainloop2(xb + (size_t)m0 * CC, wb, n0, &wl[0][0], acc);

  const int nloc   = lane & 15;
  const int rowadd = (lane >> 4) << 3;
  float bj[4];
#pragma unroll
  for (int j = 0; j < 4; ++j) bj[j] = bias[n0 + j * 16 + nloc];

#pragma unroll
  for (int mt = 0; mt < 2; ++mt) {
#pragma unroll
    for (int r = 0; r < 8; ++r) {
      int row = m0 + mt * 16 + rowadd + r;
      int t   = row & (TT - 1);
      float a0 = acc[mt][0][r] + bj[0];
      float a1 = acc[mt][1][r] + bj[1];
      if (do_rope) {
        float cz = rope[t * RR + nloc];              // cos col d == col d+16
        float sz = rope[TT * RR + t * RR + nloc];
        acc[mt][0][r] = a0 * cz - a1 * sz;
        acc[mt][1][r] = a1 * cz + a0 * sz;
      } else {
        acc[mt][0][r] = a0;
        acc[mt][1][r] = a1;
      }
      acc[mt][2][r] += bj[2];
      acc[mt][3][r] += bj[3];
    }
    if (store_mode == 0) {  // (B,H,T,D)
#pragma unroll
      for (int r = 0; r < 8; ++r) {
        int row = m0 + mt * 16 + rowadd + r;
        int bidx = row >> 11, t = row & (TT - 1);
        size_t ob = (((size_t)bidx * HH + blockIdx.y) * TT + t) * DD;
#pragma unroll
        for (int j = 0; j < 4; ++j)
          out[ob + j * 16 + nloc] = (bf16)(acc[mt][j][r] * out_scale);
      }
    } else {  // (B,H,D,T): lane holds 8 consecutive tokens for fixed dim
      int row0 = m0 + mt * 16 + rowadd;
      int bidx = row0 >> 11, t0 = row0 & (TT - 1);
#pragma unroll
      for (int j = 0; j < 4; ++j) {
        bf16 pk[8];
#pragma unroll
        for (int r = 0; r < 8; ++r) pk[r] = (bf16)(acc[mt][j][r] * out_scale);
        size_t ob =
            (((size_t)bidx * HH + blockIdx.y) * DD + j * 16 + nloc) * TT + t0;
        *(uint4*)(out + ob) = *(uint4*)pk;
      }
    }
  }
}

// ---------------------------------------------------------------------------
// Flash attention per (b, h, 128-row q block). Block 128 thr (4 waves), each
// wave owns 32 q rows (2 q-tiles sharing every K/V fragment). K tile and
// pre-transposed V^T tile both async-loaded to LDS, double-buffered -> one
// barrier per key-tile and zero VGPR staging. Online softmax runs in base-2
// (log2e folded into q projection); P restaged through per-wave LDS
// (same-wave LDS ops are in-order, no barrier needed).
// ---------------------------------------------------------------------------
__global__ __launch_bounds__(128) void attn_kernel(
    const bf16* __restrict__ q, const bf16* __restrict__ k,
    const bf16* __restrict__ vt, const int* __restrict__ mask,
    bf16* __restrict__ out) {
  __shared__ bf16 Kl[2][64 * KPITCH];     // [key][d]
  __shared__ bf16 Vt[2][64 * KPITCH];     // [d][key]
  __shared__ bf16 Pl[4][2][16 * KPITCH];  // per-wave, per-qtile P staging
  const int tid  = threadIdx.x;
  const int wave = tid >> 5, lane = tid & 31;
  const int nloc = lane & 15, rowadd = (lane >> 4) << 3;
  const int b = blockIdx.z, h = blockIdx.y;
  const size_t hoff = ((size_t)(b * HH + h)) * TT * DD;  // same for (T,D)/(D,T)
  const bf16* qh  = q + hoff;
  const bf16* kh  = k + hoff;
  const bf16* vth = vt + hoff;
  const int* maskb = mask + b * TT;
  const int q0 = blockIdx.x * 128 + wave * 32;

  v16bf aq[2][2];
#pragma unroll
  for (int qt = 0; qt < 2; ++qt) {
    aq[qt][0] = load_a_frag(qh + (size_t)(q0 + qt * 16) * DD, DD, 0);
    aq[qt][1] = load_a_frag(qh + (size_t)(q0 + qt * 16) * DD, DD, 32);
  }

  v8f o[2][4] = {{vzero8(), vzero8(), vzero8(), vzero8()},
                 {vzero8(), vzero8(), vzero8(), vzero8()}};
  float mrow[2][8], lrow[2][8];
#pragma unroll
  for (int qt = 0; qt < 2; ++qt)
#pragma unroll
    for (int r = 0; r < 8; ++r) { mrow[qt][r] = -3.0e38f; lrow[qt][r] = 0.f; }

  // prologue: issue tile 0 (async K rows [key][d], async V^T rows [d][key])
#pragma unroll
  for (int i = 0; i < 4; ++i) {
    int u = tid + i * 128;
    int row = u >> 3, c2 = u & 7;
    async_load_b128(&Kl[0][row * KPITCH + c2 * 8],
                    kh + (size_t)row * DD + c2 * 8);
    async_load_b128(&Vt[0][row * KPITCH + c2 * 8],
                    vth + (size_t)row * TT + c2 * 8);
  }

  int buf = 0;
  for (int kt = 0; kt < TT; kt += 64) {
    wait_async0();
    __syncthreads();  // tile[buf] visible to all waves

    if (kt + 64 < TT) {  // prefetch next tile into retired buffer
      int nb = buf ^ 1;
#pragma unroll
      for (int i = 0; i < 4; ++i) {
        int u = tid + i * 128;
        int row = u >> 3, c2 = u & 7;
        async_load_b128(&Kl[nb][row * KPITCH + c2 * 8],
                        kh + (size_t)(kt + 64 + row) * DD + c2 * 8);
        async_load_b128(&Vt[nb][row * KPITCH + c2 * 8],
                        vth + (size_t)row * TT + kt + 64 + c2 * 8);
      }
    }

    // S = (q * scale * log2e) K^T ; each K fragment feeds both q-tiles
    const bf16* Kb = Kl[buf];
    v8f s[2][4];
#pragma unroll
    for (int j = 0; j < 4; ++j) {
      v16bf bk0 = load_b_frag(Kb, KPITCH, j * 16, 0);
      v16bf bk1 = load_b_frag(Kb, KPITCH, j * 16, 32);
      s[0][j] = wmma_bf16(aq[0][0], bk0, vzero8());
      s[1][j] = wmma_bf16(aq[1][0], bk0, vzero8());
      s[0][j] = wmma_bf16(aq[0][1], bk1, s[0][j]);
      s[1][j] = wmma_bf16(aq[1][1], bk1, s[1][j]);
    }
    // key-padding mask (column = key, constant per lane per j-tile)
#pragma unroll
    for (int j = 0; j < 4; ++j) {
      bool valid = maskb[kt + j * 16 + nloc] != 0;
#pragma unroll
      for (int qt = 0; qt < 2; ++qt)
#pragma unroll
        for (int r = 0; r < 8; ++r)
          s[qt][j][r] = valid ? s[qt][j][r] : -1.0e30f;
    }
    // base-2 online softmax; rows live in 16-lane halves -> xor reductions
#pragma unroll
    for (int qt = 0; qt < 2; ++qt) {
#pragma unroll
      for (int r = 0; r < 8; ++r) {
        float mx = fmaxf(fmaxf(s[qt][0][r], s[qt][1][r]),
                         fmaxf(s[qt][2][r], s[qt][3][r]));
        mx = fmaxf(mx, __shfl_xor(mx, 1));
        mx = fmaxf(mx, __shfl_xor(mx, 2));
        mx = fmaxf(mx, __shfl_xor(mx, 4));
        mx = fmaxf(mx, __shfl_xor(mx, 8));
        float mnew  = fmaxf(mrow[qt][r], mx);
        float alpha = __builtin_amdgcn_exp2f(mrow[qt][r] - mnew);
        mrow[qt][r] = mnew;
        float rs = 0.f;
#pragma unroll
        for (int j = 0; j < 4; ++j) {
          float p = __builtin_amdgcn_exp2f(s[qt][j][r] - mnew);
          s[qt][j][r] = p;
          rs += p;
        }
        rs += __shfl_xor(rs, 1);
        rs += __shfl_xor(rs, 2);
        rs += __shfl_xor(rs, 4);
        rs += __shfl_xor(rs, 8);
        lrow[qt][r] = lrow[qt][r] * alpha + rs;
#pragma unroll
        for (int j = 0; j < 4; ++j) o[qt][j][r] *= alpha;
      }
      // stage P; same-wave LDS is in-order, no barrier needed
      bf16* pw = Pl[wave][qt];
#pragma unroll
      for (int j = 0; j < 4; ++j)
#pragma unroll
        for (int r = 0; r < 8; ++r)
          pw[(rowadd + r) * KPITCH + j * 16 + nloc] = (bf16)s[qt][j][r];
    }
    // O += P V ; all fragments loaded, then an 8-WMMA burst per key-half
    const bf16* Vb = Vt[buf];
#pragma unroll
    for (int ks = 0; ks < 2; ++ks) {
      v16bf ap0 = load_a_frag(Pl[wave][0], KPITCH, ks * 32);
      v16bf ap1 = load_a_frag(Pl[wave][1], KPITCH, ks * 32);
      v16bf bv0 = load_b_frag(Vb, KPITCH, 0,  ks * 32);
      v16bf bv1 = load_b_frag(Vb, KPITCH, 16, ks * 32);
      v16bf bv2 = load_b_frag(Vb, KPITCH, 32, ks * 32);
      v16bf bv3 = load_b_frag(Vb, KPITCH, 48, ks * 32);
      o[0][0] = wmma_bf16(ap0, bv0, o[0][0]);
      o[1][0] = wmma_bf16(ap1, bv0, o[1][0]);
      o[0][1] = wmma_bf16(ap0, bv1, o[0][1]);
      o[1][1] = wmma_bf16(ap1, bv1, o[1][1]);
      o[0][2] = wmma_bf16(ap0, bv2, o[0][2]);
      o[1][2] = wmma_bf16(ap1, bv2, o[1][2]);
      o[0][3] = wmma_bf16(ap0, bv3, o[0][3]);
      o[1][3] = wmma_bf16(ap1, bv3, o[1][3]);
    }
    buf ^= 1;
  }

  // normalize and write O in (B, T, H*D) layout for the output projection
#pragma unroll
  for (int qt = 0; qt < 2; ++qt)
#pragma unroll
    for (int r = 0; r < 8; ++r) {
      float inv = 1.0f / lrow[qt][r];
      int t = blockIdx.x * 128 + wave * 32 + qt * 16 + rowadd + r;
      size_t ob = ((size_t)b * TT + t) * CC + h * DD;
#pragma unroll
      for (int j = 0; j < 4; ++j)
        out[ob + j * 16 + nloc] = (bf16)(o[qt][j][r] * inv);
    }
}

// ---------------------------------------------------------------------------
// Output projection: d_out(B*T, C) f32 = attn @ Wo^T + bo
// ---------------------------------------------------------------------------
__global__ __launch_bounds__(256) void out_gemm_kernel(
    const bf16* __restrict__ ab, const bf16* __restrict__ wb,
    const float* __restrict__ bias, float* __restrict__ out) {
  __shared__ bf16 wl[2][64 * WPITCH];
  const int tid  = threadIdx.x;
  const int wave = tid >> 5, lane = tid & 31;
  const int m0 = blockIdx.x * 256 + wave * 32;
  const int n0 = blockIdx.y * 64;
  v8f acc[2][4] = {{vzero8(), vzero8(), vzero8(), vzero8()},
                   {vzero8(), vzero8(), vzero8(), vzero8()}};
  gemm_mainloop2(ab + (size_t)m0 * CC, wb, n0, &wl[0][0], acc);

  const int nloc   = lane & 15;
  const int rowadd = (lane >> 4) << 3;
  float bj[4];
#pragma unroll
  for (int j = 0; j < 4; ++j) bj[j] = bias[n0 + j * 16 + nloc];
#pragma unroll
  for (int mt = 0; mt < 2; ++mt)
#pragma unroll
    for (int r = 0; r < 8; ++r) {
      int row = m0 + mt * 16 + rowadd + r;
#pragma unroll
      for (int j = 0; j < 4; ++j)
        out[(size_t)row * CC + n0 + j * 16 + nloc] = acc[mt][j][r] + bj[j];
    }
}

// ---------------------------------------------------------------------------
extern "C" void kernel_launch(void* const* d_in, const int* in_sizes, int n_in,
                              void* d_out, int out_size, void* d_ws, size_t ws_size,
                              hipStream_t stream) {
  (void)in_sizes; (void)n_in; (void)out_size; (void)ws_size;
  const float* x    = (const float*)d_in[0];
  const float* rope = (const float*)d_in[1];
  const int*   mask = (const int*)d_in[2];
  const float* Wq   = (const float*)d_in[3];
  const float* bq   = (const float*)d_in[4];
  const float* Wk   = (const float*)d_in[5];
  const float* bk   = (const float*)d_in[6];
  const float* Wv   = (const float*)d_in[7];
  const float* bv   = (const float*)d_in[8];
  const float* Wo   = (const float*)d_in[9];
  const float* bo   = (const float*)d_in[10];
  float* out = (float*)d_out;

  char* p = (char*)d_ws;
  auto alloc = [&](size_t bytes) -> char* {
    char* r = p;
    p += (bytes + 255) & ~(size_t)255;
    return r;
  };
  bf16* xb  = (bf16*)alloc((size_t)BB * TT * CC * 2);
  bf16* wqb = (bf16*)alloc((size_t)CC * CC * 2);
  bf16* wkb = (bf16*)alloc((size_t)CC * CC * 2);
  bf16* wvb = (bf16*)alloc((size_t)CC * CC * 2);
  bf16* wob = (bf16*)alloc((size_t)CC * CC * 2);
  bf16* qw  = (bf16*)alloc((size_t)BB * HH * TT * DD * 2);
  bf16* kw  = (bf16*)alloc((size_t)BB * HH * TT * DD * 2);
  bf16* vw  = (bf16*)alloc((size_t)BB * HH * TT * DD * 2);  // (B,H,D,T)
  bf16* aw  = (bf16*)alloc((size_t)BB * TT * CC * 2);

  const int nx4 = BB * TT * CC / 4;
  const int nw4 = CC * CC / 4;
  cvt_bf16_kernel<<<(nx4 + 255) / 256, 256, 0, stream>>>(x, xb, nx4);
  cvt_bf16_kernel<<<(nw4 + 255) / 256, 256, 0, stream>>>(Wq, wqb, nw4);
  cvt_bf16_kernel<<<(nw4 + 255) / 256, 256, 0, stream>>>(Wk, wkb, nw4);
  cvt_bf16_kernel<<<(nw4 + 255) / 256, 256, 0, stream>>>(Wv, wvb, nw4);
  cvt_bf16_kernel<<<(nw4 + 255) / 256, 256, 0, stream>>>(Wo, wob, nw4);

  dim3 gG(BB * TT / 256, CC / 64);
  // 1/sqrt(D) * log2(e) folded into the q projection (softmax runs in base 2)
  qkv_gemm_kernel<<<gG, 256, 0, stream>>>(xb, wqb, bq, rope, qw, 1, 0,
                                          0.125f * 1.44269504088896f);
  qkv_gemm_kernel<<<gG, 256, 0, stream>>>(xb, wkb, bk, rope, kw, 1, 0, 1.0f);
  qkv_gemm_kernel<<<gG, 256, 0, stream>>>(xb, wvb, bv, rope, vw, 0, 1, 1.0f);

  dim3 gA(TT / 128, HH, BB);
  attn_kernel<<<gA, 128, 0, stream>>>(qw, kw, vw, mask, aw);

  out_gemm_kernel<<<gG, 256, 0, stream>>>(aw, wob, bo, out);
}
